// DynamicGraphConv_9122510536987
// MI455X (gfx1250) — matrix operand order repaired and checked by reading, not compiled, not run on hardware
//
#include <hip/hip_runtime.h>

// ---------------------------------------------------------------------------
// DynamicGraphConv for MI455X (gfx1250): bf16 WMMA pipeline.
// B=16, T=8192, F=256.
// ---------------------------------------------------------------------------

typedef __attribute__((ext_vector_type(16))) __bf16 v16bf;
typedef __attribute__((ext_vector_type(8)))  __bf16 v8bf;
typedef __attribute__((ext_vector_type(8)))  float  v8f;

static constexpr int kB = 16;
static constexpr int kT = 8192;
static constexpr int kF = 256;
static constexpr int kSlabs = 32;                 // T split for stats pass
static constexpr int kSlabT = kT / kSlabs;        // 256

// ---- workspace layout (bytes, 256-aligned) --------------------------------
static constexpr size_t OFF_YN   = 0;                                   // bf16 (B,F,T)
static constexpr size_t SZ_YN    = (size_t)kB * kF * kT * 2;            // 64 MiB
static constexpr size_t OFF_YNT  = OFF_YN + SZ_YN;                      // bf16 (B,T,F)
static constexpr size_t SZ_YNT   = SZ_YN;
static constexpr size_t OFF_ADJ  = OFF_YNT + SZ_YNT;                    // f32 (B,F,F)
static constexpr size_t SZ_ADJ   = (size_t)kB * kF * kF * 4;
static constexpr size_t OFF_ADJT = OFF_ADJ + SZ_ADJ;                    // bf16 (B,g,f) = adjS^T
static constexpr size_t SZ_ADJT  = (size_t)kB * kF * kF * 2;
static constexpr size_t OFF_MT   = OFF_ADJT + SZ_ADJT;                  // bf16 (B,f',g) = M^T
static constexpr size_t SZ_MT    = SZ_ADJT;
static constexpr size_t OFF_WT   = OFF_MT + SZ_MT;                      // bf16 (f',f) = W^T
static constexpr size_t SZ_WT    = (size_t)kF * kF * 2;
static constexpr size_t OFF_PART = OFF_WT + SZ_WT;                      // f32 pairs (B,slab,F,2)
static constexpr size_t SZ_PART  = (size_t)kB * kSlabs * kF * 2 * 4;
static constexpr size_t OFF_MEAN = OFF_PART + SZ_PART;                  // f32 (B,F)
static constexpr size_t OFF_RSTD = OFF_MEAN + (size_t)kB * kF * 4;
static constexpr size_t OFF_SIG  = OFF_RSTD + (size_t)kB * kF * 4;
static constexpr size_t OFF_C    = OFF_SIG  + (size_t)kB * kF * 4;      // f32 (B,F)
static constexpr size_t OFF_D    = OFF_C    + (size_t)kB * kF * 4;      // f32 (B,F)

// ---------------------------------------------------------------------------
// A-fragment (16x32 bf16): lane m (0..15) holds K 0..7 & 16..23; lane 16+m
// holds K 8..15 & 24..31  -> two contiguous 16B loads per lane.
// ---------------------------------------------------------------------------
__device__ __forceinline__ v16bf load_a_frag(const __bf16* p) {
    v8bf lo = *(const v8bf*)(p);
    v8bf hi = *(const v8bf*)(p + 16);
    return __builtin_shufflevector(lo, hi,
        0, 1, 2, 3, 4, 5, 6, 7, 8, 9, 10, 11, 12, 13, 14, 15);
}

__device__ __forceinline__ v8f wmma_bf16(v16bf a, v16bf b, v8f c) {
    return __builtin_amdgcn_wmma_f32_16x16x32_bf16(
        false, a, false, b, (short)0, c, false, false);
}

// Wave tile GEMM (MI*16 x NJ*16): D += A(MxK, row-major, rows m0..)
//                                    * Bt(NxK, row-major == B^T, rows n0..)
template <int MI, int NJ>
__device__ __forceinline__ void wave_gemm(
    const __bf16* __restrict__ A,  int lda, int m0,
    const __bf16* __restrict__ Bt, int ldb, int n0,
    int K, v8f (&acc)[MI][NJ])
{
    const int lane  = threadIdx.x & 31;
    const int rhalf = lane & 15;
    const int khiA  = (lane >> 4) << 3;   // 0 or 8
    const int khiB  = (lane >> 4) << 3;   // 0 or 8

    const __bf16* pa[MI];
    const __bf16* pb[NJ];
#pragma unroll
    for (int i = 0; i < MI; ++i)
        pa[i] = A + (size_t)(m0 + i * 16 + rhalf) * lda + khiA;
#pragma unroll
    for (int j = 0; j < NJ; ++j)
        pb[j] = Bt + (size_t)(n0 + j * 16 + rhalf) * ldb + khiB;

    for (int k = 0; k < K; k += 32) {
        __builtin_prefetch(pa[0] + k + 512, 0, 3);   // near-scope prefetch
        __builtin_prefetch(pb[0] + k + 512, 0, 3);
        v16bf a[MI], b[NJ];
#pragma unroll
        for (int i = 0; i < MI; ++i) a[i] = load_a_frag(pa[i] + k);
#pragma unroll
        for (int j = 0; j < NJ; ++j) b[j] = load_a_frag(pb[j] + k);
#pragma unroll
        for (int i = 0; i < MI; ++i)
#pragma unroll
            for (int j = 0; j < NJ; ++j)
                acc[i][j] = wmma_bf16(a[i], b[j], acc[i][j]);
    }
}

template <int MI, int NJ>
__device__ __forceinline__ void zero_acc(v8f (&acc)[MI][NJ]) {
#pragma unroll
    for (int i = 0; i < MI; ++i)
#pragma unroll
        for (int j = 0; j < NJ; ++j)
            acc[i][j] = (v8f){0.f, 0.f, 0.f, 0.f, 0.f, 0.f, 0.f, 0.f};
}

// ---------------------------------------------------------------------------
// K0: W (f32, FxF row-major) -> Wt bf16 (f', f)
// ---------------------------------------------------------------------------
__global__ __launch_bounds__(256) void k_wt(const float* __restrict__ W,
                                            __bf16* __restrict__ Wt) {
    int i = blockIdx.x * 256 + threadIdx.x;          // 65536 total
    int r = i >> 8, c = i & 255;
    Wt[(size_t)c * kF + r] = (__bf16)W[i];
}

// ---------------------------------------------------------------------------
// K1a: partial sum / sumsq per (b, slab, f)  — fully coalesced over f
// ---------------------------------------------------------------------------
__global__ __launch_bounds__(256) void k_stats_partial(const float* __restrict__ x,
                                                       float* __restrict__ part) {
    int bs = blockIdx.x;            // b*32 + slab
    int b  = bs >> 5, slab = bs & 31;
    int f  = threadIdx.x;
    const float* xb = x + ((size_t)b * kT + (size_t)slab * kSlabT) * kF + f;
    float s = 0.f, sq = 0.f;
    for (int t = 0; t < kSlabT; ++t) {
        float v = xb[(size_t)t * kF];
        s += v; sq += v * v;
    }
    size_t o = ((size_t)bs * kF + f) * 2;
    part[o] = s; part[o + 1] = sq;
}

// K1b: combine slabs -> mean, rstd, sigma
__global__ __launch_bounds__(256) void k_stats_final(const float* __restrict__ part,
                                                     float* __restrict__ mean,
                                                     float* __restrict__ rstd,
                                                     float* __restrict__ sigma) {
    int b = blockIdx.x, f = threadIdx.x;
    float s = 0.f, sq = 0.f;
    for (int slab = 0; slab < kSlabs; ++slab) {
        size_t o = (((size_t)b * kSlabs + slab) * kF + f) * 2;
        s += part[o]; sq += part[o + 1];
    }
    float mu  = s * (1.f / kT);
    float var = fmaxf(sq * (1.f / kT) - mu * mu, 0.f);
    float sd  = sqrtf(var) + 1e-8f;                 // STD_EPS
    mean[b * kF + f]  = mu;
    sigma[b * kF + f] = sd;
    rstd[b * kF + f]  = 1.f / sd;
}

// ---------------------------------------------------------------------------
// K2: normalize; write YnT (B,T,F) directly and Yn (B,F,T) via LDS transpose
// ---------------------------------------------------------------------------
__global__ __launch_bounds__(256) void k_normalize(const float* __restrict__ x,
                                                   const float* __restrict__ mean,
                                                   const float* __restrict__ rstd,
                                                   __bf16* __restrict__ Yn,
                                                   __bf16* __restrict__ YnT) {
    __shared__ float tile[64][65];
    int b  = blockIdx.z;
    int f0 = blockIdx.y * 64;
    int t0 = blockIdx.x * 64;

    int fl = threadIdx.x & 63;
    int tg = threadIdx.x >> 6;                      // 0..3
    float mu = mean[b * kF + f0 + fl];
    float rs = rstd[b * kF + f0 + fl];
#pragma unroll
    for (int i = 0; i < 16; ++i) {
        int tl = tg * 16 + i;
        size_t gi = ((size_t)b * kT + t0 + tl) * kF + f0 + fl;
        float y = (x[gi] - mu) * rs;
        YnT[gi] = (__bf16)y;
        tile[tl][fl] = y;
    }
    __syncthreads();
    int tl2 = threadIdx.x & 63;
    int fg  = threadIdx.x >> 6;
#pragma unroll
    for (int i = 0; i < 16; ++i) {
        int fl2 = fg * 16 + i;
        Yn[((size_t)b * kF + f0 + fl2) * kT + t0 + tl2] = (__bf16)tile[tl2][fl2];
    }
}

// ---------------------------------------------------------------------------
// K3: Gram — adj = Yn * Yn^T / T.  K = 8192, bf16 WMMA.
// grid (coltile=2, rowtile=2, b=16); block 256 = 8 waves arranged 4(m)x2(n);
// wave tile 32x64 (2 A-frags x 4 B-frags -> 8 WMMA / 12 b128 loads per k32).
// ---------------------------------------------------------------------------
__global__ __launch_bounds__(256) void k_gram(const __bf16* __restrict__ Yn,
                                              float* __restrict__ adj) {
    int b = blockIdx.z;
    int w = threadIdx.x >> 5;
    int lane = threadIdx.x & 31;
    int mbase = blockIdx.y * 128 + (w >> 1) * 32;   // f rows (4 m-groups)
    int nbase = blockIdx.x * 128 + (w & 1) * 64;    // g cols (2 n-groups)
    const __bf16* base = Yn + (size_t)b * kF * kT;

    v8f acc[2][4]; zero_acc(acc);
    wave_gemm<2, 4>(base, kT, mbase, base, kT, nbase, kT, acc);

    const float scale = 1.f / (float)kT;
    int mh = (lane >> 4) * 8;
    int n16 = lane & 15;
#pragma unroll
    for (int i = 0; i < 2; ++i)
#pragma unroll
        for (int j = 0; j < 4; ++j)
#pragma unroll
            for (int r = 0; r < 8; ++r) {
                int m = mbase + i * 16 + mh + r;
                int n = nbase + j * 16 + n16;
                adj[((size_t)b << 16) + (size_t)m * kF + n] = acc[i][j][r] * scale;
            }
}

// ---------------------------------------------------------------------------
// K3b: row softmax + fold sigma -> adjT bf16 (g,f); c_f = sum_g p*mu_g
// one wave per row (8 rows / 256-thread block)
// ---------------------------------------------------------------------------
__global__ __launch_bounds__(256) void k_softmax_fold(const float* __restrict__ adj,
                                                      const float* __restrict__ mean,
                                                      const float* __restrict__ sigma,
                                                      __bf16* __restrict__ adjT,
                                                      float* __restrict__ c) {
    int wave = threadIdx.x >> 5, lane = threadIdx.x & 31;
    int row = blockIdx.x * 8 + wave;                // 0..B*F-1
    int b = row >> 8, f = row & 255;
    const float* arow = adj + ((size_t)row << 8);

    float v[8], m = -3.4e38f;
#pragma unroll
    for (int i = 0; i < 8; ++i) { v[i] = arow[lane + 32 * i]; m = fmaxf(m, v[i]); }
#pragma unroll
    for (int off = 16; off; off >>= 1) m = fmaxf(m, __shfl_xor(m, off, 32));

    float e[8], s = 0.f;
#pragma unroll
    for (int i = 0; i < 8; ++i) { e[i] = __expf(v[i] - m); s += e[i]; }
#pragma unroll
    for (int off = 16; off; off >>= 1) s += __shfl_xor(s, off, 32);
    float inv = 1.f / s;

    float cp = 0.f;
#pragma unroll
    for (int i = 0; i < 8; ++i) {
        int g = lane + 32 * i;
        float p = e[i] * inv;
        adjT[((size_t)b << 16) + (size_t)g * kF + f] = (__bf16)(p * sigma[b * kF + g]);
        cp += p * mean[b * kF + g];
    }
#pragma unroll
    for (int off = 16; off; off >>= 1) cp += __shfl_xor(cp, off, 32);
    if (lane == 0) c[row] = cp;
}

// ---------------------------------------------------------------------------
// K4b: d[f'] = c^T W + bias
// ---------------------------------------------------------------------------
__global__ __launch_bounds__(256) void k_dvec(const float* __restrict__ c,
                                              const float* __restrict__ W,
                                              const float* __restrict__ bias,
                                              float* __restrict__ d) {
    int b = blockIdx.x, fp = threadIdx.x;
    float acc = bias[fp];
    for (int f = 0; f < kF; ++f)
        acc += c[b * kF + f] * W[(size_t)f * kF + fp];
    d[b * kF + fp] = acc;
}

// ---------------------------------------------------------------------------
// K4: M^T — Mt[f',g] = sum_f adjT[g,f] * Wt[f',f].  K=256 WMMA.
// grid (coltile=4, rowtile=2, b=16); block 256; wave tile 32x32.
// ---------------------------------------------------------------------------
__global__ __launch_bounds__(256) void k_mt(const __bf16* __restrict__ adjT,
                                            const __bf16* __restrict__ Wt,
                                            __bf16* __restrict__ Mt) {
    int b = blockIdx.z;
    int w = threadIdx.x >> 5;
    int lane = threadIdx.x & 31;
    int mbase = blockIdx.y * 128 + (w >> 1) * 32;   // g rows
    int nbase = blockIdx.x * 64  + (w & 1) * 32;    // f' cols
    const __bf16* A = adjT + ((size_t)b << 16);

    v8f acc[2][2]; zero_acc(acc);
    wave_gemm<2, 2>(A, kF, mbase, Wt, kF, nbase, kF, acc);

    int mh = (lane >> 4) * 8;
    int n16 = lane & 15;
#pragma unroll
    for (int i = 0; i < 2; ++i)
#pragma unroll
        for (int j = 0; j < 2; ++j)
#pragma unroll
            for (int r = 0; r < 8; ++r) {
                int g  = mbase + i * 16 + mh + r;
                int fp = nbase + j * 16 + n16;
                Mt[((size_t)b << 16) + (size_t)fp * kF + g] = (__bf16)acc[i][j][r];
            }
}

// ---------------------------------------------------------------------------
// K5: out = YnT @ M + d + x (residual), then LayerNorm over F.
// grid (T/32=256, B=16); block 256 = 8 waves; each wave: 32t x 32f' tile,
// workgroup covers full F=256 so LN closes in LDS.
// ---------------------------------------------------------------------------
__global__ __launch_bounds__(256) void k_out_ln(const __bf16* __restrict__ YnT,
                                                const __bf16* __restrict__ Mt,
                                                const float* __restrict__ x,
                                                const float* __restrict__ d,
                                                const float* __restrict__ gamma,
                                                const float* __restrict__ beta,
                                                float* __restrict__ out) {
    __shared__ float hbuf[32][257];
    int b  = blockIdx.y;
    int t0 = blockIdx.x * 32;
    int w  = threadIdx.x >> 5;
    int lane = threadIdx.x & 31;
    int nbase = w * 32;                              // f' slice of this wave

    const __bf16* A  = YnT + ((size_t)b * kT + t0) * kF;   // rows t, ld=F
    const __bf16* Bt = Mt + ((size_t)b << 16);             // rows f', ld=F

    v8f acc[2][2]; zero_acc(acc);
    wave_gemm<2, 2>(A, kF, 0, Bt, kF, nbase, kF, acc);

    int mh = (lane >> 4) * 8;
    int n16 = lane & 15;
#pragma unroll
    for (int i = 0; i < 2; ++i)
#pragma unroll
        for (int j = 0; j < 2; ++j)
#pragma unroll
            for (int r = 0; r < 8; ++r) {
                int tl = i * 16 + mh + r;
                int fp = nbase + j * 16 + n16;
                float h = acc[i][j][r] + d[b * kF + fp]
                        + x[((size_t)b * kT + t0 + tl) * kF + fp];
                hbuf[tl][fp] = h;
            }
    __syncthreads();

    // LayerNorm: 8 consecutive lanes per t-row (same wave), 32 f' each.
    int row = threadIdx.x >> 3;
    int sub = threadIdx.x & 7;
    float s = 0.f, sq = 0.f;
#pragma unroll
    for (int k = 0; k < 32; ++k) {
        float v = hbuf[row][sub * 32 + k];
        s += v; sq += v * v;
    }
#pragma unroll
    for (int off = 4; off; off >>= 1) {
        s  += __shfl_xor(s,  off, 32);
        sq += __shfl_xor(sq, off, 32);
    }
    float mu  = s * (1.f / kF);
    float var = fmaxf(sq * (1.f / kF) - mu * mu, 0.f);
    float rs  = rsqrtf(var + 1e-3f);                 // LN_EPS
    float* orow = out + ((size_t)b * kT + t0 + row) * kF;
#pragma unroll
    for (int k = 0; k < 32; ++k) {
        int fp = sub * 32 + k;
        orow[fp] = gamma[fp] * (hbuf[row][fp] - mu) * rs + beta[fp];
    }
}

// ---------------------------------------------------------------------------
extern "C" void kernel_launch(void* const* d_in, const int* in_sizes, int n_in,
                              void* d_out, int out_size, void* d_ws, size_t ws_size,
                              hipStream_t stream) {
    const float* x     = (const float*)d_in[0];
    const float* W     = (const float*)d_in[1];
    const float* bias  = (const float*)d_in[2];
    const float* gamma = (const float*)d_in[3];
    const float* beta  = (const float*)d_in[4];
    float* out = (float*)d_out;

    char* ws = (char*)d_ws;
    __bf16* Yn   = (__bf16*)(ws + OFF_YN);
    __bf16* YnT  = (__bf16*)(ws + OFF_YNT);
    float*  adj  = (float*)(ws + OFF_ADJ);
    __bf16* adjT = (__bf16*)(ws + OFF_ADJT);
    __bf16* Mt   = (__bf16*)(ws + OFF_MT);
    __bf16* Wt   = (__bf16*)(ws + OFF_WT);
    float*  part = (float*)(ws + OFF_PART);
    float*  mean = (float*)(ws + OFF_MEAN);
    float*  rstd = (float*)(ws + OFF_RSTD);
    float*  sig  = (float*)(ws + OFF_SIG);
    float*  cvec = (float*)(ws + OFF_C);
    float*  dvec = (float*)(ws + OFF_D);

    k_wt<<<kF * kF / 256, 256, 0, stream>>>(W, Wt);
    k_stats_partial<<<kB * kSlabs, 256, 0, stream>>>(x, part);
    k_stats_final<<<kB, 256, 0, stream>>>(part, mean, rstd, sig);
    k_normalize<<<dim3(kT / 64, kF / 64, kB), 256, 0, stream>>>(x, mean, rstd, Yn, YnT);
    k_gram<<<dim3(kF / 128, kF / 128, kB), 256, 0, stream>>>(Yn, adj);
    k_softmax_fold<<<kB * kF / 8, 256, 0, stream>>>(adj, mean, sig, adjT, cvec);
    k_dvec<<<kB, 256, 0, stream>>>(cvec, W, bias, dvec);
    k_mt<<<dim3(kF / 64, kF / 128, kB), 256, 0, stream>>>(adjT, Wt, Mt);
    k_out_ln<<<dim3(kT / 32, kB), 256, 0, stream>>>(YnT, Mt, x, dvec, gamma, beta, out);
}
